// GBottleneck_24111946400391
// MI455X (gfx1250) — matrix operand order repaired and checked by reading, not compiled, not run on hardware
//
#include <hip/hip_runtime.h>
#include <hip/hip_bf16.h>

// ---------------- types ----------------
typedef __attribute__((ext_vector_type(16))) __bf16 v16bf;
typedef __attribute__((ext_vector_type(8)))  float  v8f;
typedef unsigned short ushort_t;
typedef unsigned int   uint_t;

union V16 {
    v16bf  v;
    uint4  q[2];
    ushort_t u[16];
};

#define GN      50000
#define GE      300000
#define GDIN    963
#define GH      256
#define GDOUT   3
#define GNBLK   6

// float -> bf16 (round to nearest even), bit manipulation only
__device__ __forceinline__ ushort_t f2bf(float f) {
    unsigned int u = __float_as_uint(f);
    unsigned int r = u + 0x7FFFu + ((u >> 16) & 1u);
    return (ushort_t)(r >> 16);
}

__device__ __forceinline__ void atomic_add_f32(float* p, float v) {
    __hip_atomic_fetch_add(p, v, __ATOMIC_RELAXED, __HIP_MEMORY_SCOPE_AGENT);
}

// ---------------------------------------------------------------------------
// Pack a weight matrix W[K][256] (f32, row-major) into WMMA B-fragment order:
//   out[((kc*16 + ct)*32 + lane)*16 + e] = bf16( W[kc*32 + (lane>>4)*16 + e]
//                                                 [ct*16 + (lane&15)] )
// One thread per (kc, ct, lane); each writes one contiguous 32-byte fragment
// slice, so the GEMM can fetch B fragments with two global_load_b128.
// ---------------------------------------------------------------------------
__global__ void pack_weights(const float* __restrict__ W,
                             ushort_t* __restrict__ out, int K)
{
    int idx = blockIdx.x * blockDim.x + threadIdx.x;
    int nkc = (K + 31) / 32;
    int total = nkc * 16 * 32;
    if (idx >= total) return;

    int lane = idx & 31;
    int ct   = (idx >> 5) & 15;
    int kc   = idx >> 9;
    int n     = ct * 16 + (lane & 15);
    int kbase = kc * 32 + (lane >> 4) * 16;

    V16 t;
#pragma unroll
    for (int e = 0; e < 16; ++e) {
        int k = kbase + e;
        float v = (k < K) ? W[(size_t)k * GH + n] : 0.f;
        t.u[e] = f2bf(v);
    }
    uint4* dst = (uint4*)(out + (size_t)idx * 16);
    dst[0] = t.q[0];
    dst[1] = t.q[1];
}

// ---------------------------------------------------------------------------
// Fused dual GEMM (bf16 WMMA, f32 accumulate):
//   supOut[r, :] = A[r, :] @ W          (N x H)
//   accOut[r, :] = A[r, :] @ Wl + bias  (N x H)
// Workgroup: 256 threads = 8 waves. Tile: 32 rows x 256 cols (H).
// Wave w: row-group rg = w&1 (16 rows), col-quarter cq = w>>1 (4 tiles of 16).
// A is staged into LDS in fragment order (ds_store_b64 / ds_load_b128);
// B fragments come pre-packed from global (2x global_load_b128 each).
// ---------------------------------------------------------------------------
__global__ void __launch_bounds__(256)
gemm_dual_bf16(const float* __restrict__ A,
               const ushort_t* __restrict__ Wp,
               const ushort_t* __restrict__ Wlp,
               const float* __restrict__ bias,
               float* __restrict__ supOut,
               float* __restrict__ accOut,
               int nRows, int K)
{
    // A tile in fragment order: [rg(2)][lane(32)][elem(16)] bf16 = 2 KB
    __shared__ ushort_t AsF[2 * 32 * 16];

    const int tid  = threadIdx.x;
    const int wid  = tid >> 5;
    const int lane = tid & 31;
    const int rg   = wid & 1;        // row group (0/1) -> 16 rows each
    const int cq   = wid >> 1;       // col quarter (0..3) -> 4 tiles of 16
    const int rowBase = blockIdx.x * 32;

    const int lo = lane & 15;
    const int hi = lane >> 4;

    // staging decomposition (4 bf16 elements per thread)
    const int t4     = tid * 4;
    const int rg_s   = t4 >> 9;          // 0..1
    const int lane_s = (t4 >> 4) & 31;   // target fragment lane
    const int e0     = t4 & 15;          // 0,4,8,12
    const int hs     = lane_s >> 4;
    // fragment elem e -> K = hs*16+e
    const int k_l    = hs * 16 + e0;
    const int row_s  = rowBase + rg_s * 16 + (lane_s & 15);
    const int ldsOff = rg_s * 512 + lane_s * 16 + e0;   // ushort units

    v8f accW[4];
    v8f accL[4];
    const v8f vzero = {0.f, 0.f, 0.f, 0.f, 0.f, 0.f, 0.f, 0.f};
#pragma unroll
    for (int i = 0; i < 4; ++i) { accW[i] = vzero; accL[i] = vzero; }

    const int nkc = (K + 31) / 32;
    for (int kc = 0; kc < nkc; ++kc) {
        const int kb = kc * 32;
        __syncthreads();
        // ---- stage A tile (fragment order): 4 f32 loads + 1 ds_store_b64 ----
        {
            float f[4];
#pragma unroll
            for (int j = 0; j < 4; ++j) {
                int k = kb + k_l + j;
                f[j] = (row_s < nRows && k < K) ? A[(size_t)row_s * K + k] : 0.f;
            }
            uint_t w0 = (uint_t)f2bf(f[0]) | ((uint_t)f2bf(f[1]) << 16);
            uint_t w1 = (uint_t)f2bf(f[2]) | ((uint_t)f2bf(f[3]) << 16);
            *(uint2*)&AsF[ldsOff] = make_uint2(w0, w1);
        }
        __syncthreads();

        // ---- A fragment: two ds_load_b128 ----
        V16 af;
        {
            const uint4* pa = (const uint4*)&AsF[rg * 512 + lane * 16];
            af.q[0] = pa[0];
            af.q[1] = pa[1];
        }

        // ---- 4 column tiles x 2 matrices; B frags direct from global ----
#pragma unroll
        for (int ct = 0; ct < 4; ++ct) {
            size_t fbase = (((size_t)kc * 16 + (cq * 4 + ct)) * 32 + lane) * 16;
            V16 bw, bl;
            const uint4* pw = (const uint4*)(Wp  + fbase);
            const uint4* pl = (const uint4*)(Wlp + fbase);
            bw.q[0] = pw[0]; bw.q[1] = pw[1];
            bl.q[0] = pl[0]; bl.q[1] = pl[1];
            accW[ct] = __builtin_amdgcn_wmma_f32_16x16x32_bf16(
                false, af.v, false, bw.v, (short)0, accW[ct], false, false);
            accL[ct] = __builtin_amdgcn_wmma_f32_16x16x32_bf16(
                false, af.v, false, bl.v, (short)0, accL[ct], false, false);
        }
    }

    // ---- write out (C/D layout: row = rg*16 + hi*8 + i, col = lane&15) ----
#pragma unroll
    for (int ct = 0; ct < 4; ++ct) {
        int col = (cq * 4 + ct) * 16 + lo;
        float bv = bias[col];
#pragma unroll
        for (int i = 0; i < 8; ++i) {
            int row = rowBase + rg * 16 + hi * 8 + i;
            if (row < nRows) {
                size_t o = (size_t)row * GH + col;
                supOut[o] = accW[ct][i];
                accOut[o] = accL[ct][i] + bv;
            }
        }
    }
}

// ---------------------------------------------------------------------------
// Edge aggregation: acc[dst, c] += ew * sup[src, c].  One block per edge,
// one lane per channel; hardware f32 atomics (relaxed, device scope).
// ---------------------------------------------------------------------------
__global__ void __launch_bounds__(GH)
scatter_edges(const float* __restrict__ sup,
              const int* __restrict__ src,
              const int* __restrict__ dst,
              const float* __restrict__ ew,
              float* __restrict__ acc)
{
    int e = blockIdx.x;
    int c = threadIdx.x;
    int s = src[e];
    int d = dst[e];
    float w = ew[e];
    atomic_add_f32(&acc[(size_t)d * GH + c], w * sup[(size_t)s * GH + c]);
}

__global__ void relu_kernel(const float* __restrict__ in,
                            float* __restrict__ out, int n)
{
    int i = blockIdx.x * blockDim.x + threadIdx.x;
    if (i < n) out[i] = fmaxf(in[i], 0.f);
}

// h = (h + relu(acc)) * 0.5
__global__ void residual_kernel(float* __restrict__ h,
                                const float* __restrict__ acc, int n)
{
    int i = blockIdx.x * blockDim.x + threadIdx.x;
    if (i < n) h[i] = (h[i] + fmaxf(acc[i], 0.f)) * 0.5f;
}

// ---------------------------------------------------------------------------
// Output head (H=256 -> DOUT=3): per-node dot products.
// ---------------------------------------------------------------------------
__global__ void head_dense(const float* __restrict__ h,
                           const float* __restrict__ W2,
                           const float* __restrict__ Wl2,
                           const float* __restrict__ b2,
                           float* __restrict__ sup2,
                           float* __restrict__ xout)
{
    int n = blockIdx.x * blockDim.x + threadIdx.x;
    if (n >= GN) return;
    float aS[GDOUT] = {0.f, 0.f, 0.f};
    float aL[GDOUT] = {0.f, 0.f, 0.f};
    const float* hr = h + (size_t)n * GH;
    for (int k = 0; k < GH; ++k) {
        float hv = hr[k];
#pragma unroll
        for (int j = 0; j < GDOUT; ++j) {
            aS[j] = fmaf(hv, W2 [k * GDOUT + j], aS[j]);
            aL[j] = fmaf(hv, Wl2[k * GDOUT + j], aL[j]);
        }
    }
#pragma unroll
    for (int j = 0; j < GDOUT; ++j) {
        sup2[(size_t)n * GDOUT + j] = aS[j];
        xout[(size_t)n * GDOUT + j] = aL[j] + b2[j];
    }
}

__global__ void head_scatter(const float* __restrict__ sup2,
                             const int* __restrict__ src,
                             const int* __restrict__ dst,
                             const float* __restrict__ ew,
                             float* __restrict__ xout)
{
    int e = blockIdx.x * blockDim.x + threadIdx.x;
    if (e >= GE) return;
    int s = src[e];
    int d = dst[e];
    float w = ew[e];
#pragma unroll
    for (int j = 0; j < GDOUT; ++j)
        atomic_add_f32(&xout[(size_t)d * GDOUT + j],
                       w * sup2[(size_t)s * GDOUT + j]);
}

// ---------------------------------------------------------------------------
extern "C" void kernel_launch(void* const* d_in, const int* in_sizes, int n_in,
                              void* d_out, int out_size, void* d_ws, size_t ws_size,
                              hipStream_t stream)
{
    (void)in_sizes; (void)n_in; (void)out_size; (void)ws_size;

    const float* x    = (const float*)d_in[0];
    const int*   src  = (const int*)  d_in[1];
    const int*   dst  = (const int*)  d_in[2];
    const float* ew   = (const float*)d_in[3];
    const float* W1   = (const float*)d_in[4];
    const float* Wl1  = (const float*)d_in[5];
    const float* b1   = (const float*)d_in[6];
    const float* blkW = (const float*)d_in[7];   // [6,2,H,H]
    const float* blkWl= (const float*)d_in[8];
    const float* blkb = (const float*)d_in[9];   // [6,2,H]
    const float* W2   = (const float*)d_in[10];
    const float* Wl2  = (const float*)d_in[11];
    const float* b2   = (const float*)d_in[12];

    float* xout   = (float*)d_out;               // [N, 3]
    float* hidout = xout + (size_t)GN * GDOUT;   // [N, H]

    const size_t NH = (size_t)GN * GH;
    float* sup  = (float*)d_ws;        // N x H
    float* acc  = sup  + NH;           // N x H
    float* h    = acc  + NH;           // N x H
    float* y    = h    + NH;           // N x H
    float* sup2 = y    + NH;           // N x 3

    // packed-weight region (bf16 fragment order), 16B-aligned by construction
    ushort_t* packBase = (ushort_t*)(sup2 + (size_t)GN * GDOUT);
    const size_t packL1  = (size_t)((GDIN + 31) / 32) * 16 * 32 * 16; // per matrix
    const size_t packBlk = (size_t)((GH   + 31) / 32) * 16 * 32 * 16; // per matrix
    ushort_t* W1p  = packBase;
    ushort_t* Wl1p = W1p + packL1;
    ushort_t* blkP = Wl1p + packL1;    // 24 matrices: [(b*2+c)*2 + {0:W,1:Wl}]

    const dim3 blk256(256);
    const dim3 gemmGrid((GN + 31) / 32);
    const dim3 ewGrid((int)((NH + 255) / 256));

    // ---- pack all weights into fragment order (cheap, L2-resident) ----
    {
        int tL1  = ((GDIN + 31) / 32) * 512;
        int tBlk = ((GH + 31) / 32) * 512;
        pack_weights<<<dim3((tL1 + 255) / 256), blk256, 0, stream>>>(W1,  W1p,  GDIN);
        pack_weights<<<dim3((tL1 + 255) / 256), blk256, 0, stream>>>(Wl1, Wl1p, GDIN);
        for (int i = 0; i < GNBLK * 2; ++i) {
            pack_weights<<<dim3((tBlk + 255) / 256), blk256, 0, stream>>>(
                blkW  + (size_t)i * GH * GH, blkP + (size_t)(2 * i + 0) * packBlk, GH);
            pack_weights<<<dim3((tBlk + 255) / 256), blk256, 0, stream>>>(
                blkWl + (size_t)i * GH * GH, blkP + (size_t)(2 * i + 1) * packBlk, GH);
        }
    }

    // ---- layer 1: DIN=963 -> H ----
    gemm_dual_bf16<<<gemmGrid, blk256, 0, stream>>>(x, W1p, Wl1p, b1, sup, acc, GN, GDIN);
    scatter_edges <<<dim3(GE), dim3(GH), 0, stream>>>(sup, src, dst, ew, acc);
    relu_kernel   <<<ewGrid, blk256, 0, stream>>>(acc, h, (int)NH);

    // ---- 6 residual blocks ----
    for (int b = 0; b < GNBLK; ++b) {
        for (int c = 0; c < 2; ++c) {
            int wi = b * 2 + c;
            const ushort_t* Wp  = blkP + (size_t)(2 * wi + 0) * packBlk;
            const ushort_t* Wlp = blkP + (size_t)(2 * wi + 1) * packBlk;
            const float*    bp  = blkb + (size_t)wi * GH;
            const float*    in  = (c == 0) ? h : y;
            gemm_dual_bf16<<<gemmGrid, blk256, 0, stream>>>(in, Wp, Wlp, bp, sup, acc, GN, GH);
            scatter_edges <<<dim3(GE), dim3(GH), 0, stream>>>(sup, src, dst, ew, acc);
            if (c == 0)
                relu_kernel    <<<ewGrid, blk256, 0, stream>>>(acc, y, (int)NH);
            else
                residual_kernel<<<ewGrid, blk256, 0, stream>>>(h, acc, (int)NH);
        }
    }

    // ---- output head: H -> 3, plus hidden-state copy ----
    head_dense  <<<dim3((GN + 255) / 256), blk256, 0, stream>>>(h, W2, Wl2, b2, sup2, xout);
    head_scatter<<<dim3((GE + 255) / 256), blk256, 0, stream>>>(sup2, src, dst, ew, xout);
    hipMemcpyAsync(hidout, h, NH * sizeof(float), hipMemcpyDeviceToDevice, stream);
}
